// GCN_32650341384807
// MI455X (gfx1250) — matrix operand order repaired and checked by reading, not compiled, not run on hardware
//
#include <hip/hip_runtime.h>
#include <hip/hip_bf16.h>

// ---------------------------------------------------------------------------
// GCN (3x GCNConv + mean-pool + linear head) for MI455X (gfx1250, wave32).
//
// Perf model: edge scatter-add dominates (~2.5 GB of L2-resident atomic
// traffic across 3 layers; agg buffer = 12.8 MB, fully L2-resident at 192 MB).
// Dense GEMMs run on v_wmma_f32_16x16x32_bf16 with weights pre-packed into
// B-fragment order (bf16) so the inner loop does pure b128 loads + WMMA.
// ---------------------------------------------------------------------------

#define IN_CH    128
#define HID      32
#define OUT_CH   10
#define NG       512

typedef __attribute__((ext_vector_type(16))) __bf16 bf16x16;
typedef __attribute__((ext_vector_type(8)))  float  f32x8;

// ---------------------------------------------------------------------------
// Degree accumulation: deg[dst] += 1 for every edge (self-loop added later).
// ---------------------------------------------------------------------------
__global__ void deg_kernel(const int* __restrict__ dst, float* __restrict__ deg,
                           int nedges) {
    int e = blockIdx.x * blockDim.x + threadIdx.x;
    if (e < nedges) atomicAdd(&deg[dst[e]], 1.0f);
}

// deg -> dinv = rsqrt(deg + 1)   (self-loop guarantees deg+1 >= 1)
__global__ void dinv_kernel(float* __restrict__ deg, int n) {
    int i = blockIdx.x * blockDim.x + threadIdx.x;
    if (i < n) deg[i] = rsqrtf(deg[i] + 1.0f);
}

// ---------------------------------------------------------------------------
// Pack W[K x 32] (f32, row-major) into bf16 WMMA B-fragment order:
//   Wp[ ((kt*2 + nt)*32 + lane)*16 + j ] = W[(kt*32 + half*8 + j + (j>=8 ? 8 : 0))*32 + nt*16 + n]
// with half = lane>>4, n = lane&15.  One thread per packed element.
// ---------------------------------------------------------------------------
__global__ void pack_w(const float* __restrict__ W, __bf16* __restrict__ Wp,
                       int total) {
    int idx = blockIdx.x * blockDim.x + threadIdx.x;
    if (idx >= total) return;
    int j    = idx & 15;
    int lane = (idx >> 4) & 31;
    int nt   = (idx >> 9) & 1;
    int kt   = idx >> 10;
    int half = lane >> 4;
    int n    = lane & 15;
    Wp[idx] = (__bf16)W[(size_t)(kt * 32 + half * 8 + j + (j >= 8 ? 8 : 0)) * HID + nt * 16 + n];
}

// ---------------------------------------------------------------------------
// out[N x 32] = A[N x K] * W[K x 32], bf16 inputs, f32 accumulate via WMMA.
// One wave handles a 16-row x 32-col tile (2 n-tiles, K/32 k-steps).
// B comes pre-packed: one 32-byte contiguous vector load per fragment.
// No branches around WMMA: OOB rows clamp loads, mask stores (EXEC all-1s).
// ---------------------------------------------------------------------------
template <int K>
__global__ void gemm_wmma_bf16(const float* __restrict__ A,
                               const __bf16* __restrict__ Wp,
                               float* __restrict__ out, int nrows) {
    const int lane = threadIdx.x & 31;
    const int wave = (blockIdx.x * blockDim.x + threadIdx.x) >> 5;
    const int half = lane >> 4;      // 0: lanes 0-15, 1: lanes 16-31
    const int m    = lane & 15;      // row within tile / col within tile
    const int row0 = wave * 16;

    int row = row0 + m;
    if (row >= nrows) row = nrows - 1;   // clamp loads; stores masked below

    f32x8 acc0 = {};
    f32x8 acc1 = {};

    #pragma unroll
    for (int kt = 0; kt < K / 32; ++kt) {
        const int k0 = kt * 32;
        // A fragment, 16x32 bf16 (ISA 7.12.2): lane half selects K sub-ranges.
        bf16x16 a;
        const float* ap = A + (size_t)row * K + k0 + half * 8;
        #pragma unroll
        for (int j = 0; j < 8; ++j) a[j]     = (__bf16)ap[j];       // K = k0+half*8+j
        #pragma unroll
        for (int j = 0; j < 8; ++j) a[8 + j] = (__bf16)ap[16 + j];  // K = k0+16+half*8+j

        // B fragments: contiguous 32B per lane, pre-packed & pre-converted.
        bf16x16 b0 = *(const bf16x16*)(Wp + ((size_t)(kt * 2 + 0) * 32 + lane) * 16);
        bf16x16 b1 = *(const bf16x16*)(Wp + ((size_t)(kt * 2 + 1) * 32 + lane) * 16);

        acc0 = __builtin_amdgcn_wmma_f32_16x16x32_bf16(
            false, a, false, b0, (short)0, acc0, false, false);
        acc1 = __builtin_amdgcn_wmma_f32_16x16x32_bf16(
            false, a, false, b1, (short)0, acc1, false, false);
    }

    // D layout: VGPR r holds M = r + 8*half, N = lane&15.
    #pragma unroll
    for (int r = 0; r < 8; ++r) {
        int rr = row0 + r + 8 * half;
        if (rr < nrows) {
            out[(size_t)rr * HID + m]      = acc0[r];
            out[(size_t)rr * HID + 16 + m] = acc1[r];
        }
    }
}

// ---------------------------------------------------------------------------
// Edge scatter: agg[dst] += h[src] * dinv[src]*dinv[dst].
// One wave per edge, one lane per feature: src/dst/dinv are wave-broadcast,
// h gather + atomic add are single coalesced 128B transactions (L2-resident).
// ---------------------------------------------------------------------------
__global__ void edge_scatter(const int* __restrict__ src,
                             const int* __restrict__ dst,
                             const float* __restrict__ dinv,
                             const float* __restrict__ h,
                             float* __restrict__ agg, int nedges) {
    long long gid = (long long)blockIdx.x * blockDim.x + threadIdx.x;
    int e = (int)(gid >> 5);
    int f = (int)(gid & 31);
    if (e >= nedges) return;
    int s = src[e], d = dst[e];
    float norm = dinv[s] * dinv[d];
    atomicAdd(&agg[(size_t)d * HID + f], h[(size_t)s * HID + f] * norm);
}

// Self-loop contribution + bias + optional ReLU, in place on agg.
__global__ void finalize_kernel(const float* __restrict__ h,
                                const float* __restrict__ dinv,
                                const float* __restrict__ bias,
                                float* __restrict__ agg, int nnodes, int relu) {
    long long gid = (long long)blockIdx.x * blockDim.x + threadIdx.x;
    int i = (int)(gid >> 5);
    int f = (int)(gid & 31);
    if (i >= nnodes) return;
    float di = dinv[i];
    float v = agg[(size_t)i * HID + f] + h[(size_t)i * HID + f] * di * di + bias[f];
    if (relu) v = fmaxf(v, 0.0f);
    agg[(size_t)i * HID + f] = v;
}

// Global mean pool (accumulate): sums[batch[i]] += h[i]; cnt[batch[i]] += 1.
__global__ void pool_accum(const float* __restrict__ h,
                           const int* __restrict__ batch,
                           float* __restrict__ sums, float* __restrict__ cnt,
                           int nnodes) {
    long long gid = (long long)blockIdx.x * blockDim.x + threadIdx.x;
    int i = (int)(gid >> 5);
    int f = (int)(gid & 31);
    if (i >= nnodes) return;
    int g = batch[i];
    atomicAdd(&sums[(size_t)g * HID + f], h[(size_t)i * HID + f]);
    if (f == 0) atomicAdd(&cnt[g], 1.0f);
}

// Head: out[g][o] = (sums[g]/max(cnt,1)) . Wl[:,o] + bl[o]   (512x10, tiny)
__global__ void head_kernel(const float* __restrict__ sums,
                            const float* __restrict__ cnt,
                            const float* __restrict__ Wl,
                            const float* __restrict__ bl,
                            float* __restrict__ out) {
    int gid = blockIdx.x * blockDim.x + threadIdx.x;
    if (gid >= NG * OUT_CH) return;
    int g = gid / OUT_CH, o = gid % OUT_CH;
    float inv = 1.0f / fmaxf(cnt[g], 1.0f);
    float acc = bl[o];
    #pragma unroll
    for (int f = 0; f < HID; ++f)
        acc += (sums[g * HID + f] * inv) * Wl[f * OUT_CH + o];
    out[gid] = acc;
}

// ---------------------------------------------------------------------------
// Launcher. Inputs (setup_inputs order):
//  0:x [N,128] f32   1:edge_index [2,E] i32   2:batch [N] i32
//  3:W1[128,32] 4:b1[32] 5:W2[32,32] 6:b2[32] 7:W3[32,32] 8:b3[32]
//  9:Wl[32,10] 10:bl[10]        out: [512,10] f32
// ---------------------------------------------------------------------------
extern "C" void kernel_launch(void* const* d_in, const int* in_sizes, int n_in,
                              void* d_out, int out_size, void* d_ws, size_t ws_size,
                              hipStream_t stream) {
    const float* x     = (const float*)d_in[0];
    const int*   eidx  = (const int*)d_in[1];
    const int*   batch = (const int*)d_in[2];
    const float* W1 = (const float*)d_in[3];
    const float* b1 = (const float*)d_in[4];
    const float* W2 = (const float*)d_in[5];
    const float* b2 = (const float*)d_in[6];
    const float* W3 = (const float*)d_in[7];
    const float* b3 = (const float*)d_in[8];
    const float* Wl = (const float*)d_in[9];
    const float* bl = (const float*)d_in[10];

    const int n_nodes = in_sizes[0] / IN_CH;   // 100000
    const int n_edges = in_sizes[1] / 2;       // 3200000
    const int* src = eidx;                     // edge_index[0]
    const int* dst = eidx + n_edges;           // edge_index[1]

    // Workspace carve-up. Packed weights first (fixed 12 KB, keeps the
    // 32-byte B-fragment alignment independent of n_nodes), then f32 arrays.
    const int W1P_ELEMS = (IN_CH / 32) * 2 * 32 * 16;   // 4096
    const int WHP_ELEMS = (HID   / 32) * 2 * 32 * 16;   // 1024
    __bf16* W1p = (__bf16*)d_ws;                        // 4096 bf16
    __bf16* W2p = W1p + W1P_ELEMS;                      // 1024 bf16
    __bf16* W3p = W2p + WHP_ELEMS;                      // 1024 bf16
    float* dinv = (float*)((char*)d_ws + 12288);        // N
    float* bufA = dinv + n_nodes;                       // N*32
    float* bufB = bufA + (size_t)n_nodes * HID;         // N*32
    float* psum = bufB + (size_t)n_nodes * HID;         // 512*32
    float* pcnt = psum + (size_t)NG * HID;              // 512

    const int BT = 256;
    const int gEdge  = (int)(((long long)n_edges * 32 + BT - 1) / BT);
    const int gNode  = (int)(((long long)n_nodes * 32 + BT - 1) / BT);
    const int gGemm  = (n_nodes + 127) / 128;           // 8 waves/blk * 16 rows
    const int g1d_n  = (n_nodes + BT - 1) / BT;
    const int g1d_e  = (n_edges + BT - 1) / BT;
    const size_t nhid_bytes = (size_t)n_nodes * HID * sizeof(float);

    // --- pack weights to bf16 B-fragment order ---
    pack_w<<<(W1P_ELEMS + BT - 1) / BT, BT, 0, stream>>>(W1, W1p, W1P_ELEMS);
    pack_w<<<(WHP_ELEMS + BT - 1) / BT, BT, 0, stream>>>(W2, W2p, WHP_ELEMS);
    pack_w<<<(WHP_ELEMS + BT - 1) / BT, BT, 0, stream>>>(W3, W3p, WHP_ELEMS);

    // --- symmetric norm: dinv = rsqrt(deg + 1) ---
    hipMemsetAsync(dinv, 0, (size_t)n_nodes * sizeof(float), stream);
    deg_kernel<<<g1d_e, BT, 0, stream>>>(dst, dinv, n_edges);
    dinv_kernel<<<g1d_n, BT, 0, stream>>>(dinv, n_nodes);

    // --- layer 1: h = X@W1 (WMMA), aggregate, +b1, ReLU ---
    gemm_wmma_bf16<IN_CH><<<gGemm, BT, 0, stream>>>(x, W1p, bufA, n_nodes);
    hipMemsetAsync(bufB, 0, nhid_bytes, stream);
    edge_scatter<<<gEdge, BT, 0, stream>>>(src, dst, dinv, bufA, bufB, n_edges);
    finalize_kernel<<<gNode, BT, 0, stream>>>(bufA, dinv, b1, bufB, n_nodes, 1);

    // --- layer 2 ---
    gemm_wmma_bf16<HID><<<gGemm, BT, 0, stream>>>(bufB, W2p, bufA, n_nodes);
    hipMemsetAsync(bufB, 0, nhid_bytes, stream);
    edge_scatter<<<gEdge, BT, 0, stream>>>(src, dst, dinv, bufA, bufB, n_edges);
    finalize_kernel<<<gNode, BT, 0, stream>>>(bufA, dinv, b2, bufB, n_nodes, 1);

    // --- layer 3 (no ReLU) ---
    gemm_wmma_bf16<HID><<<gGemm, BT, 0, stream>>>(bufB, W3p, bufA, n_nodes);
    hipMemsetAsync(bufB, 0, nhid_bytes, stream);
    edge_scatter<<<gEdge, BT, 0, stream>>>(src, dst, dinv, bufA, bufB, n_edges);
    finalize_kernel<<<gNode, BT, 0, stream>>>(bufA, dinv, b3, bufB, n_nodes, 0);

    // --- mean pool + head ---
    hipMemsetAsync(psum, 0, (size_t)NG * (HID + 1) * sizeof(float), stream);
    pool_accum<<<gNode, BT, 0, stream>>>(bufB, batch, psum, pcnt, n_nodes);
    head_kernel<<<(NG * OUT_CH + BT - 1) / BT, BT, 0, stream>>>(
        psum, pcnt, Wl, bl, (float*)d_out);
}
